// Discriminator_16423954940199
// MI455X (gfx1250) — matrix operand order repaired and checked by reading, not compiled, not run on hardware
//
#include <hip/hip_runtime.h>
#include <math.h>

#define T_STEPS 512
#define BATCH   64
#define DIM     1024
#define NDIM    4096   // 4*DIM

typedef __attribute__((ext_vector_type(16))) __bf16 bf16x16;
typedef __attribute__((ext_vector_type(8)))  float  f32x8;

union Frag16 {
  bf16x16 v;
  unsigned short s[16];
  uint4 q[2];
};

__device__ __forceinline__ unsigned short f2bf(float f) {
  union { float f; unsigned u; } c; c.f = f;
  unsigned u = c.u;
  u += 0x7FFFu + ((u >> 16) & 1u);   // round-to-nearest-even
  return (unsigned short)(u >> 16);
}
__device__ __forceinline__ float bf2f(unsigned short h) {
  union { unsigned u; float f; } c; c.u = ((unsigned)h) << 16;
  return c.f;
}
__device__ __forceinline__ float sigmoidf(float x) {
  return 1.0f / (1.0f + expf(-x));
}

// Load a 16-bit A/B fragment whose two 8-element chunks sit at `base` and
// `base+16` elements within a row of pre-split bf16 data.
__device__ __forceinline__ bf16x16 load_frag_split(const unsigned short* row, int k0) {
  Frag16 f;
  f.q[0] = *(const uint4*)(row + k0);
  f.q[1] = *(const uint4*)(row + k0 + 16);
  return f.v;
}

// ---------------------------------------------------------------------------
// Split fp32 weights into bf16 hi/lo pairs (row-major [4D, D], same layout).
// ---------------------------------------------------------------------------
__global__ void prep_weights(const float* __restrict__ Wih,
                             const float* __restrict__ Whh,
                             unsigned short* __restrict__ wih_hi,
                             unsigned short* __restrict__ wih_lo,
                             unsigned short* __restrict__ whh_hi,
                             unsigned short* __restrict__ whh_lo) {
  int i = blockIdx.x * blockDim.x + threadIdx.x;
  if (i < NDIM * DIM) {
    float w = Wih[i];
    unsigned short h = f2bf(w);
    wih_hi[i] = h;
    wih_lo[i] = f2bf(w - bf2f(h));
    w = Whh[i];
    h = f2bf(w);
    whh_hi[i] = h;
    whh_lo[i] = f2bf(w - bf2f(h));
  }
}

// ---------------------------------------------------------------------------
// One-shot split of x[T*B, D] fp32 into bf16 hi/lo planes (each element
// converted exactly once, instead of 256x redundantly inside the GEMM).
// ---------------------------------------------------------------------------
__global__ void split_x(const float* __restrict__ x,
                        unsigned short* __restrict__ xhi,
                        unsigned short* __restrict__ xlo) {
  int i = blockIdx.x * blockDim.x + threadIdx.x;
  if (i < T_STEPS * BATCH * DIM) {
    float f = x[i];
    unsigned short h = f2bf(f);
    xhi[i] = h;
    xlo[i] = f2bf(f - bf2f(h));
  }
}

__global__ void init_state(float* __restrict__ h, float* __restrict__ c,
                           unsigned short* __restrict__ hhi,
                           unsigned short* __restrict__ hlo) {
  int i = blockIdx.x * blockDim.x + threadIdx.x;
  if (i < BATCH * DIM) {
    h[i] = 0.0f; c[i] = 0.0f; hhi[i] = 0; hlo[i] = 0;
  }
}

// ---------------------------------------------------------------------------
// Hoisted input GEMM: Gx[t*B + b, n] = x[t,b,:]·W_ih[n,:] + b_ih[n] + b_hh[n]
// One wave per 16x16 tile; 2048 M-tiles x 256 N-tiles = 524288 waves.
// x already pre-split to bf16 hi/lo => inner loop is pure loads + 3 WMMAs.
// ---------------------------------------------------------------------------
__global__ void __launch_bounds__(128)
gx_gemm(const unsigned short* __restrict__ xhi,
        const unsigned short* __restrict__ xlo,
        const unsigned short* __restrict__ wih_hi,
        const unsigned short* __restrict__ wih_lo,
        const float* __restrict__ b_ih,
        const float* __restrict__ b_hh,
        float* __restrict__ gx) {
  const int lane  = threadIdx.x & 31;
  const int wave  = (blockIdx.x << 2) | (threadIdx.x >> 5);
  const int nt    = wave & 255;     // N tile
  const int mt    = wave >> 8;      // M tile (0..2047)
  const int ln    = lane & 15;
  const int hi16  = lane >> 4;
  const int abase = hi16 << 3;      // A-frag K sub-offset (0/8)
  const int bbase = hi16 << 4;      // unused; B-frag loads also use abase (0/8)

  const unsigned short* xrow_hi = xhi + (size_t)(mt * 16 + ln) * DIM;
  const unsigned short* xrow_lo = xlo + (size_t)(mt * 16 + ln) * DIM;
  const int nc = nt * 16 + ln;
  const unsigned short* bhrow = wih_hi + (size_t)nc * DIM;
  const unsigned short* blrow = wih_lo + (size_t)nc * DIM;

  f32x8 acc = {};
#pragma unroll 1
  for (int kb = 0; kb < DIM; kb += 32) {
    bf16x16 ah = load_frag_split(xrow_hi, kb + abase);
    bf16x16 al = load_frag_split(xrow_lo, kb + abase);
    bf16x16 bh = load_frag_split(bhrow, kb + abase);
    bf16x16 bl = load_frag_split(blrow, kb + abase);
    acc = __builtin_amdgcn_wmma_f32_16x16x32_bf16(false, ah, false, bh, (short)0, acc, false, false);
    acc = __builtin_amdgcn_wmma_f32_16x16x32_bf16(false, ah, false, bl, (short)0, acc, false, false);
    acc = __builtin_amdgcn_wmma_f32_16x16x32_bf16(false, al, false, bh, (short)0, acc, false, false);
  }
  const float bias  = b_ih[nc] + b_hh[nc];
  const int   rbase = mt * 16 + (hi16 << 3);
#pragma unroll
  for (int r = 0; r < 8; ++r)
    gx[(size_t)(rbase + r) * NDIM + nt * 16 + ln] = acc[r] + bias;
}

// ---------------------------------------------------------------------------
// One LSTM time step, fully fused: wave owns M-tile mt (16 batch rows) and
// the i/f/g/o gate columns for one 16-wide d slice, so it can do the cell
// update in-registers. h is double-buffered (bf16 hi/lo) across launches.
// MODE 0: fused x-GEMM, x converted on the fly (no extra scratch)
// MODE 1: fused x-GEMM, x pre-split to bf16 hi/lo
// MODE 2: x contribution read from precomputed Gx (half the serial WMMA work)
// ---------------------------------------------------------------------------
template <int MODE>
__global__ void __launch_bounds__(128)
lstm_step(const float* __restrict__ x,
          const unsigned short* __restrict__ xhi,
          const unsigned short* __restrict__ xlo,
          const float* __restrict__ gx,
          const unsigned short* __restrict__ wih_hi,
          const unsigned short* __restrict__ wih_lo,
          const unsigned short* __restrict__ whh_hi,
          const unsigned short* __restrict__ whh_lo,
          const float* __restrict__ b_ih,
          const float* __restrict__ b_hh,
          const unsigned short* __restrict__ hin_hi,
          const unsigned short* __restrict__ hin_lo,
          unsigned short* __restrict__ hout_hi,
          unsigned short* __restrict__ hout_lo,
          float* __restrict__ c_state,
          float* __restrict__ h_f32,
          int t) {
  const int lane  = threadIdx.x & 31;
  const int wave  = (blockIdx.x << 2) | (threadIdx.x >> 5);
  const int mt    = wave >> 6;      // 0..3  (batch tile)
  const int j     = wave & 63;      // 0..63 (d tile within each gate)
  const int ln    = lane & 15;
  const int hi16  = lane >> 4;
  const int abase = hi16 << 3;
  const int bbase = hi16 << 4;

  const int arow = mt * 16 + ln;
  const unsigned short* hrow_hi = hin_hi + (size_t)arow * DIM;
  const unsigned short* hrow_lo = hin_lo + (size_t)arow * DIM;
  const float* xrow = x + ((size_t)t * BATCH + arow) * DIM;
  const unsigned short* xrow_hi = xhi + ((size_t)t * BATCH + arow) * DIM;
  const unsigned short* xrow_lo = xlo + ((size_t)t * BATCH + arow) * DIM;

  const unsigned short* whh_hi_row[4];
  const unsigned short* whh_lo_row[4];
  const unsigned short* wih_hi_row[4];
  const unsigned short* wih_lo_row[4];
#pragma unroll
  for (int g = 0; g < 4; ++g) {
    const int nc = g * DIM + j * 16 + ln;
    whh_hi_row[g] = whh_hi + (size_t)nc * DIM;
    whh_lo_row[g] = whh_lo + (size_t)nc * DIM;
    wih_hi_row[g] = wih_hi + (size_t)nc * DIM;
    wih_lo_row[g] = wih_lo + (size_t)nc * DIM;
  }

  const int rbase = mt * 16 + (hi16 << 3);

  f32x8 acc[4];
#pragma unroll
  for (int g = 0; g < 4; ++g) {
    const int nc = g * DIM + j * 16 + ln;
    if constexpr (MODE == 2) {
#pragma unroll
      for (int r = 0; r < 8; ++r)
        acc[g][r] = gx[((size_t)t * BATCH + rbase + r) * NDIM + nc];
    } else {
      const float bias = b_ih[nc] + b_hh[nc];
#pragma unroll
      for (int r = 0; r < 8; ++r) acc[g][r] = bias;
    }
  }

#pragma unroll 1
  for (int kb = 0; kb < DIM; kb += 32) {
    bf16x16 hh = load_frag_split(hrow_hi, kb + abase);
    bf16x16 hl = load_frag_split(hrow_lo, kb + abase);

    bf16x16 xhf, xlf;
    if constexpr (MODE == 1) {
      xhf = load_frag_split(xrow_hi, kb + abase);
      xlf = load_frag_split(xrow_lo, kb + abase);
    } else if constexpr (MODE == 0) {
      Frag16 fh, fl;
#pragma unroll
      for (int half = 0; half < 2; ++half) {
        const float* p = xrow + kb + abase + half * 16;
        float4 f0 = *(const float4*)(p);
        float4 f1 = *(const float4*)(p + 4);
        float fv[8] = {f0.x, f0.y, f0.z, f0.w, f1.x, f1.y, f1.z, f1.w};
#pragma unroll
        for (int e = 0; e < 8; ++e) {
          unsigned short h = f2bf(fv[e]);
          fh.s[half * 8 + e] = h;
          fl.s[half * 8 + e] = f2bf(fv[e] - bf2f(h));
        }
      }
      xhf = fh.v; xlf = fl.v;
    }

#pragma unroll
    for (int g = 0; g < 4; ++g) {
      bf16x16 bh = load_frag_split(whh_hi_row[g], kb + abase);
      bf16x16 bl = load_frag_split(whh_lo_row[g], kb + abase);
      acc[g] = __builtin_amdgcn_wmma_f32_16x16x32_bf16(false, hh, false, bh, (short)0, acc[g], false, false);
      acc[g] = __builtin_amdgcn_wmma_f32_16x16x32_bf16(false, hh, false, bl, (short)0, acc[g], false, false);
      acc[g] = __builtin_amdgcn_wmma_f32_16x16x32_bf16(false, hl, false, bh, (short)0, acc[g], false, false);
      if constexpr (MODE != 2) {
        bf16x16 wbh = load_frag_split(wih_hi_row[g], kb + abase);
        bf16x16 wbl = load_frag_split(wih_lo_row[g], kb + abase);
        acc[g] = __builtin_amdgcn_wmma_f32_16x16x32_bf16(false, xhf, false, wbh, (short)0, acc[g], false, false);
        acc[g] = __builtin_amdgcn_wmma_f32_16x16x32_bf16(false, xhf, false, wbl, (short)0, acc[g], false, false);
        acc[g] = __builtin_amdgcn_wmma_f32_16x16x32_bf16(false, xlf, false, wbh, (short)0, acc[g], false, false);
      }
    }
  }

  // Fused cell update: this wave exclusively owns (rows rbase..rbase+7, col d).
  const int d = j * 16 + ln;
#pragma unroll
  for (int r = 0; r < 8; ++r) {
    const int row = rbase + r;
    const size_t off = (size_t)row * DIM + d;
    float iv = sigmoidf(acc[0][r]);
    float fv = sigmoidf(acc[1][r]);
    float gv = tanhf(acc[2][r]);
    float ov = sigmoidf(acc[3][r]);
    float cn = fv * c_state[off] + iv * gv;
    float hn = ov * tanhf(cn);
    c_state[off] = cn;
    h_f32[off] = hn;
    unsigned short hb = f2bf(hn);
    hout_hi[off] = hb;
    hout_lo[off] = f2bf(hn - bf2f(hb));
  }
}

// ---------------------------------------------------------------------------
// logit = log_softmax(h_final @ W_proj^T + b_proj)   [64, 2] — tiny, scalar.
// ---------------------------------------------------------------------------
__global__ void proj_logsoftmax(const float* __restrict__ h,
                                const float* __restrict__ Wp,
                                const float* __restrict__ bp,
                                float* __restrict__ out) {
  int b = threadIdx.x;
  if (b >= BATCH) return;
  float z0 = bp[0], z1 = bp[1];
  for (int k = 0; k < DIM; ++k) {
    float hv = h[(size_t)b * DIM + k];
    z0 = fmaf(hv, Wp[k], z0);
    z1 = fmaf(hv, Wp[DIM + k], z1);
  }
  float m = fmaxf(z0, z1);
  float l = m + logf(expf(z0 - m) + expf(z1 - m));
  out[2 * b + 0] = z0 - l;
  out[2 * b + 1] = z1 - l;
}

extern "C" void kernel_launch(void* const* d_in, const int* in_sizes, int n_in,
                              void* d_out, int out_size, void* d_ws, size_t ws_size,
                              hipStream_t stream) {
  (void)in_sizes; (void)n_in; (void)out_size;
  const float* x   = (const float*)d_in[0];
  const float* Wih = (const float*)d_in[1];
  const float* Whh = (const float*)d_in[2];
  const float* bih = (const float*)d_in[3];
  const float* bhh = (const float*)d_in[4];
  const float* Wp  = (const float*)d_in[5];
  const float* bp  = (const float*)d_in[6];
  float* out = (float*)d_out;

  char* ws = (char*)d_ws;
  const size_t WBYTES = (size_t)NDIM * DIM * sizeof(unsigned short); // 8 MB each
  unsigned short* wih_hi = (unsigned short*)(ws);
  unsigned short* wih_lo = (unsigned short*)(ws + WBYTES);
  unsigned short* whh_hi = (unsigned short*)(ws + 2 * WBYTES);
  unsigned short* whh_lo = (unsigned short*)(ws + 3 * WBYTES);
  size_t off = 4 * WBYTES;
  float* h_f32 = (float*)(ws + off); off += (size_t)BATCH * DIM * sizeof(float);
  float* c_st  = (float*)(ws + off); off += (size_t)BATCH * DIM * sizeof(float);
  unsigned short* hhi[2]; unsigned short* hlo[2];
  hhi[0] = (unsigned short*)(ws + off); off += (size_t)BATCH * DIM * 2;
  hlo[0] = (unsigned short*)(ws + off); off += (size_t)BATCH * DIM * 2;
  hhi[1] = (unsigned short*)(ws + off); off += (size_t)BATCH * DIM * 2;
  hlo[1] = (unsigned short*)(ws + off); off += (size_t)BATCH * DIM * 2;
  const size_t base_need = off;                                       // 33 MB
  const size_t xsplit_bytes = (size_t)T_STEPS * BATCH * DIM * 2;      // 64 MB each
  unsigned short* x_hi = (unsigned short*)(ws + off);
  unsigned short* x_lo = (unsigned short*)(ws + off + xsplit_bytes);
  const size_t xsplit_end = base_need + 2 * xsplit_bytes;             // +128 MB
  const size_t gx_bytes = (size_t)T_STEPS * BATCH * NDIM * sizeof(float); // 512 MB
  float* gx = (float*)(ws + xsplit_end);

  if (ws_size < base_need) return;  // scratch too small to run at all
  const int mode = (ws_size >= xsplit_end + gx_bytes) ? 2
                 : (ws_size >= xsplit_end)            ? 1
                                                      : 0;

  prep_weights<<<(NDIM * DIM + 255) / 256, 256, 0, stream>>>(
      Wih, Whh, wih_hi, wih_lo, whh_hi, whh_lo);
  init_state<<<(BATCH * DIM + 255) / 256, 256, 0, stream>>>(
      h_f32, c_st, hhi[0], hlo[0]);

  if (mode >= 1) {
    split_x<<<(T_STEPS * BATCH * DIM + 255) / 256, 256, 0, stream>>>(x, x_hi, x_lo);
  }
  if (mode == 2) {
    // 2048 M-tiles * 256 N-tiles waves, 4 waves per 128-thread block.
    gx_gemm<<<(2048 * 256) / 4, 128, 0, stream>>>(
        x_hi, x_lo, wih_hi, wih_lo, bih, bhh, gx);
  }

  for (int t = 0; t < T_STEPS; ++t) {
    const int pin = t & 1, pout = pin ^ 1;
    if (mode == 2) {
      lstm_step<2><<<64, 128, 0, stream>>>(
          x, x_hi, x_lo, gx, wih_hi, wih_lo, whh_hi, whh_lo, bih, bhh,
          hhi[pin], hlo[pin], hhi[pout], hlo[pout], c_st, h_f32, t);
    } else if (mode == 1) {
      lstm_step<1><<<64, 128, 0, stream>>>(
          x, x_hi, x_lo, gx, wih_hi, wih_lo, whh_hi, whh_lo, bih, bhh,
          hhi[pin], hlo[pin], hhi[pout], hlo[pout], c_st, h_f32, t);
    } else {
      lstm_step<0><<<64, 128, 0, stream>>>(
          x, x_hi, x_lo, gx, wih_hi, wih_lo, whh_hi, whh_lo, bih, bhh,
          hhi[pin], hlo[pin], hhi[pout], hlo[pout], c_st, h_f32, t);
    }
  }

  proj_logsoftmax<<<1, 64, 0, stream>>>(h_f32, Wp, bp, out);
}
